// SimVIModule_28338194219615
// MI455X (gfx1250) — compile-verified, matrix operand order checked by reading, not run on hardware
//
#include <hip/hip_runtime.h>
#include <hip/hip_bf16.h>
#include <math.h>

#define NN 30000
#define GG 2000
#define HH 128
#define ZZ 20
#define SS 10
#define EE 960000
#define NTILES 1875   // NN/16

typedef _Float16 half16 __attribute__((ext_vector_type(16)));
typedef float    float8 __attribute__((ext_vector_type(8)));

// Fast transcendentals: v_log_f32 / v_exp_f32 hardware paths.
__device__ __forceinline__ float fast_log1p(float v) { return __logf(1.0f + v); }
__device__ __forceinline__ float fast_exp(float v)   { return __expf(v); }

// ---------------------------------------------------------------------------
// WMMA helpers (gfx1250, wave32).  Layouts per CDNA5 ISA 7.12.2:
//  A (16x32 f16): lane<16 -> row=lane,   K = {0..7, 16..23}
//                 lane>=16 -> row=lane-16, K = {8..15, 24..31}
//  B (32x16 f16): mirrored: col = lane&15, same K striping per lane group.
//  C/D (16x16 f32, 8 VGPRs): row = r + 8*(lane>=16), col = lane&15.
// ---------------------------------------------------------------------------
__device__ __forceinline__ float8 wmma_f16(half16 a, half16 b, float8 c) {
  return __builtin_amdgcn_wmma_f32_16x16x32_f16(false, a, false, b, (short)0, c,
                                                false, false);
}

// Load a 16x32 A tile from an f32 row-major matrix, converting to f16.
// Optionally apply log1p and accumulate the raw sum (for library).
template <bool DO_LOG1P, bool DO_SUM>
__device__ __forceinline__ half16 a_tile_full(const float* __restrict__ mat,
                                              int ld, int r0, int k0,
                                              float& sum) {
  const int lane = threadIdx.x & 31;
  const float* rowp = mat + (size_t)(r0 + (lane & 15)) * ld;
  const int base = k0 + ((lane & 16) ? 8 : 0);
  half16 a;
#pragma unroll
  for (int g = 0; g < 2; ++g) {
    const float4 u = *reinterpret_cast<const float4*>(rowp + base + g * 16);
    const float4 w = *reinterpret_cast<const float4*>(rowp + base + g * 16 + 4);
    float vals[8] = {u.x, u.y, u.z, u.w, w.x, w.y, w.z, w.w};
#pragma unroll
    for (int j = 0; j < 8; ++j) {
      float v = vals[j];
      if (DO_SUM) sum += v;
      if (DO_LOG1P) v = fast_log1p(v);
      a[g * 8 + j] = (_Float16)v;
    }
  }
  return a;
}

template <bool DO_LOG1P, bool DO_SUM>
__device__ __forceinline__ half16 a_tile_tail(const float* __restrict__ mat,
                                              int ld, int r0, int k0, int kmax,
                                              float& sum) {
  const int lane = threadIdx.x & 31;
  const float* rowp = mat + (size_t)(r0 + (lane & 15)) * ld;
  const int base = k0 + ((lane & 16) ? 8 : 0);
  half16 a;
#pragma unroll
  for (int g = 0; g < 2; ++g) {
#pragma unroll
    for (int j = 0; j < 8; ++j) {
      const int k = base + g * 16 + j;
      float v = (k < kmax) ? rowp[k] : 0.0f;
      if (DO_SUM) sum += v;
      if (DO_LOG1P) v = fast_log1p(v);
      a[g * 8 + j] = (_Float16)v;
    }
  }
  return a;
}

__device__ __forceinline__ half16 b_tile(const _Float16* __restrict__ pk,
                                         int tileIdx) {
  const int lane = threadIdx.x & 31;
  return *reinterpret_cast<const half16*>(pk + ((size_t)tileIdx * 32 + lane) * 16);
}

__device__ __forceinline__ void store_c(float* __restrict__ out, int ld, int r0,
                                        int n0, const float8& acc,
                                        const float* __restrict__ bias,
                                        bool relu) {
  const int lane = threadIdx.x & 31;
  const int col = n0 + (lane & 15);
  const int rbase = r0 + ((lane & 16) ? 8 : 0);
  const float b = bias ? bias[col] : 0.0f;
#pragma unroll
  for (int r = 0; r < 8; ++r) {
    float v = acc[r] + b;
    if (relu) v = fmaxf(v, 0.0f);
    out[(size_t)(rbase + r) * ld + col] = v;
  }
}

// ---------------------------------------------------------------------------
// Pack an f32 row-major (K x Ncols) weight into f16 WMMA B tiles.
// dst[t] with t = ((kc*NT + nt)*32 + lane)*16 + h holds W[k(lane,h)][n(lane)].
// ---------------------------------------------------------------------------
__global__ void pack_b_kernel(const float* __restrict__ W,
                              _Float16* __restrict__ dst, int K, int Ncols,
                              int NT, int total) {
  const int t = blockIdx.x * blockDim.x + threadIdx.x;
  if (t >= total) return;
  const int h = t & 15;
  const int lane = (t >> 4) & 31;
  const int tile = t >> 9;
  const int kc = tile / NT;
  const int nt = tile % NT;
  const int grp = (lane >> 4) & 1;
  const int kl = (h < 8) ? (h + grp * 8) : (16 + (h - 8) + grp * 8);
  const int k = kc * 32 + kl;
  const int n = nt * 16 + (lane & 15);
  const float v = (k < K && n < Ncols) ? W[(size_t)k * Ncols + n] : 0.0f;
  dst[t] = (_Float16)v;
}

// ---------------------------------------------------------------------------
// Encoder layer 0:  h1 = relu(log1p(x) @ W0 + b0),  libsum = rowsum(x)
// One wave per 16-row tile.  K = 2000 -> 62 full kchunks + 1 tail.
// ---------------------------------------------------------------------------
__global__ __launch_bounds__(128) void enc0_kernel(
    const float* __restrict__ x, const _Float16* __restrict__ pkW0,
    const float* __restrict__ b0, float* __restrict__ h1,
    float* __restrict__ libsum) {
  const int tile = blockIdx.x * 4 + (threadIdx.x >> 5);
  if (tile >= NTILES) return;
  const int r0 = tile * 16;
  const int lane = threadIdx.x & 31;
  float8 acc[8] = {};
  float rowsum = 0.0f;
  const float* rowp = x + (size_t)(r0 + (lane & 15)) * GG;

  for (int kc = 0; kc < 62; ++kc) {
    if (kc + 2 < 62) __builtin_prefetch(rowp + (kc + 2) * 32, 0, 0);
    half16 a = a_tile_full<true, true>(x, GG, r0, kc * 32, rowsum);
#pragma unroll
    for (int nt = 0; nt < 8; ++nt)
      acc[nt] = wmma_f16(a, b_tile(pkW0, kc * 8 + nt), acc[nt]);
  }
  {  // tail: k = 1984..1999 (zero padded to 2016)
    half16 a = a_tile_tail<true, true>(x, GG, r0, 62 * 32, GG, rowsum);
#pragma unroll
    for (int nt = 0; nt < 8; ++nt)
      acc[nt] = wmma_f16(a, b_tile(pkW0, 62 * 8 + nt), acc[nt]);
  }
  // library = log(rowsum); we keep the raw sum since only exp(library) is used.
  const float tot = rowsum + __shfl_xor(rowsum, 16, 32);
  if (lane < 16) libsum[r0 + lane] = tot;
#pragma unroll
  for (int nt = 0; nt < 8; ++nt)
    store_c(h1, HH, r0, nt * 16, acc[nt], b0, true);
}

// Encoder layer 1:  h2 = relu(h1 @ W1 + b1)
__global__ __launch_bounds__(128) void enc1_kernel(
    const float* __restrict__ h1, const _Float16* __restrict__ pkW1,
    const float* __restrict__ b1, float* __restrict__ h2) {
  const int tile = blockIdx.x * 4 + (threadIdx.x >> 5);
  if (tile >= NTILES) return;
  const int r0 = tile * 16;
  float8 acc[8] = {};
  float dummy = 0.0f;
#pragma unroll
  for (int kc = 0; kc < 4; ++kc) {
    half16 a = a_tile_full<false, false>(h1, HH, r0, kc * 32, dummy);
#pragma unroll
    for (int nt = 0; nt < 8; ++nt)
      acc[nt] = wmma_f16(a, b_tile(pkW1, kc * 8 + nt), acc[nt]);
  }
#pragma unroll
  for (int nt = 0; nt < 8; ++nt)
    store_c(h2, HH, r0, nt * 16, acc[nt], b1, true);
}

// ---------------------------------------------------------------------------
// VAE head: q_m = h2@Wm+bm, q_v = exp(h2@Wv+bv)+eps, z = q_m + sqrt(q_v)*eps_z,
// feat = q_m[:, 10:20].  One wave per node; shuffle reductions over H=128.
// ---------------------------------------------------------------------------
__global__ __launch_bounds__(128) void vae_head_kernel(
    const float* __restrict__ h2, const float* __restrict__ Wm,
    const float* __restrict__ bm, const float* __restrict__ Wv,
    const float* __restrict__ bv, const float* __restrict__ eps_z,
    float* __restrict__ z, float* __restrict__ feat) {
  const int n = blockIdx.x * 4 + (threadIdx.x >> 5);
  if (n >= NN) return;
  const int lane = threadIdx.x & 31;
  float hv[4];
#pragma unroll
  for (int q = 0; q < 4; ++q) hv[q] = h2[(size_t)n * HH + lane + 32 * q];
  for (int c = 0; c < ZZ; ++c) {
    float pm = 0.0f, pv = 0.0f;
#pragma unroll
    for (int q = 0; q < 4; ++q) {
      const int k = lane + 32 * q;
      pm += hv[q] * Wm[k * ZZ + c];
      pv += hv[q] * Wv[k * ZZ + c];
    }
#pragma unroll
    for (int o = 16; o > 0; o >>= 1) {
      pm += __shfl_xor(pm, o, 32);
      pv += __shfl_xor(pv, o, 32);
    }
    if (lane == 0) {
      const float qm = pm + bm[c];
      const float qv = fast_exp(pv + bv[c]) + 1e-4f;
      z[(size_t)n * ZZ + c] = qm + sqrtf(qv) * eps_z[(size_t)n * ZZ + c];
      if (c >= SS) feat[(size_t)n * SS + (c - SS)] = qm;
    }
  }
}

// GATv2 node linear transforms: xl/xr for both mean and var branches.
__global__ void gat_linear_kernel(
    const float* __restrict__ feat, const float* __restrict__ Wl_m,
    const float* __restrict__ bl_m, const float* __restrict__ Wr_m,
    const float* __restrict__ br_m, const float* __restrict__ Wl_v,
    const float* __restrict__ bl_v, const float* __restrict__ Wr_v,
    const float* __restrict__ br_v, float* __restrict__ xlm,
    float* __restrict__ xrm, float* __restrict__ xlv,
    float* __restrict__ xrv) {
  const int n = blockIdx.x * blockDim.x + threadIdx.x;
  if (n >= NN) return;
  float f[SS];
#pragma unroll
  for (int k = 0; k < SS; ++k) f[k] = feat[(size_t)n * SS + k];
#pragma unroll
  for (int j = 0; j < SS; ++j) {
    float a = bl_m[j], b = br_m[j], c = bl_v[j], d = br_v[j];
#pragma unroll
    for (int k = 0; k < SS; ++k) {
      a += f[k] * Wl_m[k * SS + j];
      b += f[k] * Wr_m[k * SS + j];
      c += f[k] * Wl_v[k * SS + j];
      d += f[k] * Wr_v[k * SS + j];
    }
    xlm[(size_t)n * SS + j] = a;
    xrm[(size_t)n * SS + j] = b;
    xlv[(size_t)n * SS + j] = c;
    xrv[(size_t)n * SS + j] = d;
  }
}

__global__ void gat_init_kernel(float* __restrict__ m_m, float* __restrict__ m_v,
                                float* __restrict__ s_m, float* __restrict__ s_v,
                                float* __restrict__ acc_m,
                                float* __restrict__ acc_v) {
  const int i = blockIdx.x * blockDim.x + threadIdx.x;
  if (i < NN) {
    const float ninf = __uint_as_float(0xFF800000u);
    m_m[i] = ninf; m_v[i] = ninf;
    s_m[i] = 0.0f; s_v[i] = 0.0f;
  }
  if (i < NN * SS) { acc_m[i] = 0.0f; acc_v[i] = 0.0f; }
}

__device__ __forceinline__ void atomicMaxFloat(float* addr, float v) {
  if (v >= 0.0f)
    atomicMax(reinterpret_cast<int*>(addr), __float_as_int(v));
  else
    atomicMin(reinterpret_cast<unsigned int*>(addr), __float_as_uint(v));
}

// Edge pass 1: attention logits + segment max (float atomic-max via int trick).
__global__ void edge_score_kernel(
    const int* __restrict__ src, const int* __restrict__ dst,
    const float* __restrict__ xlm, const float* __restrict__ xrm,
    const float* __restrict__ xlv, const float* __restrict__ xrv,
    const float* __restrict__ att_m, const float* __restrict__ att_v,
    float* __restrict__ e_m, float* __restrict__ e_v, float* __restrict__ m_m,
    float* __restrict__ m_v) {
  const int e = blockIdx.x * blockDim.x + threadIdx.x;
  if (e >= EE) return;
  const int s = src[e], d = dst[e];
  float em = 0.0f, ev = 0.0f;
#pragma unroll
  for (int j = 0; j < SS; ++j) {
    float a = xlm[(size_t)s * SS + j] + xrm[(size_t)d * SS + j];
    a = (a > 0.0f) ? a : 0.2f * a;
    em += a * att_m[j];
    float b = xlv[(size_t)s * SS + j] + xrv[(size_t)d * SS + j];
    b = (b > 0.0f) ? b : 0.2f * b;
    ev += b * att_v[j];
  }
  e_m[e] = em; e_v[e] = ev;
  atomicMaxFloat(m_m + d, em);
  atomicMaxFloat(m_v + d, ev);
}

// Edge pass 2: ee = exp(e - m[dst]); segment-sum ee.
__global__ void edge_exp_kernel(const int* __restrict__ dst,
                                float* __restrict__ e_m, float* __restrict__ e_v,
                                const float* __restrict__ m_m,
                                const float* __restrict__ m_v,
                                float* __restrict__ s_m,
                                float* __restrict__ s_v) {
  const int e = blockIdx.x * blockDim.x + threadIdx.x;
  if (e >= EE) return;
  const int d = dst[e];
  const float em = fast_exp(e_m[e] - m_m[d]);
  const float ev = fast_exp(e_v[e] - m_v[d]);
  e_m[e] = em; e_v[e] = ev;
  atomicAdd(s_m + d, em);
  atomicAdd(s_v + d, ev);
}

// Edge pass 3: out[dst] += alpha * xl[src].
__global__ void edge_accum_kernel(
    const int* __restrict__ src, const int* __restrict__ dst,
    const float* __restrict__ e_m, const float* __restrict__ e_v,
    const float* __restrict__ s_m, const float* __restrict__ s_v,
    const float* __restrict__ xlm, const float* __restrict__ xlv,
    float* __restrict__ acc_m, float* __restrict__ acc_v) {
  const int e = blockIdx.x * blockDim.x + threadIdx.x;
  if (e >= EE) return;
  const int s = src[e], d = dst[e];
  const float am = e_m[e] / (s_m[d] + 1e-16f);
  const float av = e_v[e] / (s_v[d] + 1e-16f);
#pragma unroll
  for (int j = 0; j < SS; ++j) {
    atomicAdd(&acc_m[(size_t)d * SS + j], am * xlm[(size_t)s * SS + j]);
    atomicAdd(&acc_v[(size_t)d * SS + j], av * xlv[(size_t)s * SS + j]);
  }
}

// Node head: z_gat, z_all, hd = relu(layernorm(z_all @ dec_W0 + b0)).
__global__ __launch_bounds__(128) void node_head_kernel(
    const float* __restrict__ acc_m, const float* __restrict__ acc_v,
    const float* __restrict__ gm_bias, const float* __restrict__ gv_bias,
    const float* __restrict__ eps_gat, const float* __restrict__ z,
    const float* __restrict__ decW0, const float* __restrict__ decb0,
    float* __restrict__ hd) {
  const int n = blockIdx.x * 4 + (threadIdx.x >> 5);
  if (n >= NN) return;
  const int lane = threadIdx.x & 31;
  float za[ZZ + SS];
#pragma unroll
  for (int j = 0; j < SS; ++j) {
    const float qm = acc_m[(size_t)n * SS + j] + gm_bias[j];
    const float qv = fast_exp(acc_v[(size_t)n * SS + j] + gv_bias[j]) + 1e-4f;
    za[j] = qm + sqrtf(qv) * eps_gat[(size_t)n * SS + j];
  }
#pragma unroll
  for (int k = 0; k < ZZ; ++k) za[SS + k] = z[(size_t)n * ZZ + k];
  float h[4], s1 = 0.0f, s2 = 0.0f;
#pragma unroll
  for (int q = 0; q < 4; ++q) {
    const int c = lane + 32 * q;
    float v = decb0[c];
#pragma unroll
    for (int k = 0; k < ZZ + SS; ++k) v += za[k] * decW0[k * HH + c];
    h[q] = v; s1 += v; s2 += v * v;
  }
#pragma unroll
  for (int o = 16; o > 0; o >>= 1) {
    s1 += __shfl_xor(s1, o, 32);
    s2 += __shfl_xor(s2, o, 32);
  }
  const float mean = s1 * (1.0f / HH);
  const float var = s2 * (1.0f / HH) - mean * mean;
  const float inv = rsqrtf(var + 1e-5f);
#pragma unroll
  for (int q = 0; q < 4; ++q) {
    float v = (h[q] - mean) * inv;
    hd[(size_t)n * HH + lane + 32 * q] = fmaxf(v, 0.0f);
  }
}

// Decoder GEMM: logits = hd @ Ws + bs -> d_out (softmaxed in-place next).
// grid.y splits the 125 N-tiles into 5 chunks of 25.
__global__ __launch_bounds__(128) void dec_gemm_kernel(
    const float* __restrict__ hd, const _Float16* __restrict__ pkWs,
    const float* __restrict__ bs, float* __restrict__ out) {
  const int tile = blockIdx.x * 4 + (threadIdx.x >> 5);
  if (tile >= NTILES) return;
  const int r0 = tile * 16;
  float dummy = 0.0f;
  half16 aReg[4];
#pragma unroll
  for (int kc = 0; kc < 4; ++kc)
    aReg[kc] = a_tile_full<false, false>(hd, HH, r0, kc * 32, dummy);
  const int nt0 = blockIdx.y * 25;
  for (int t = 0; t < 25; ++t) {
    const int nt = nt0 + t;
    float8 acc = {};
#pragma unroll
    for (int kc = 0; kc < 4; ++kc)
      acc = wmma_f16(aReg[kc], b_tile(pkWs, kc * 125 + nt), acc);
    store_c(out, GG, r0, nt * 16, acc, bs, false);
  }
}

// Row softmax over G=2000 + scale by rowsum(x) (== exp(library)): in-place.
__global__ __launch_bounds__(256) void softmax_kernel(
    float* __restrict__ out, const float* __restrict__ libsum) {
  __shared__ float red[256];
  const int row = blockIdx.x;
  const int tid = threadIdx.x;
  float* p = out + (size_t)row * GG;
  float mx = -INFINITY;
  for (int c = tid; c < GG; c += 256) mx = fmaxf(mx, p[c]);
  red[tid] = mx; __syncthreads();
  for (int s = 128; s > 0; s >>= 1) {
    if (tid < s) red[tid] = fmaxf(red[tid], red[tid + s]);
    __syncthreads();
  }
  mx = red[0]; __syncthreads();
  float sm = 0.0f;
  for (int c = tid; c < GG; c += 256) sm += fast_exp(p[c] - mx);
  red[tid] = sm; __syncthreads();
  for (int s = 128; s > 0; s >>= 1) {
    if (tid < s) red[tid] += red[tid + s];
    __syncthreads();
  }
  const float scale = libsum[row] / red[0];
  for (int c = tid; c < GG; c += 256) p[c] = fast_exp(p[c] - mx) * scale;
}

// ---------------------------------------------------------------------------
extern "C" void kernel_launch(void* const* d_in, const int* in_sizes, int n_in,
                              void* d_out, int out_size, void* d_ws,
                              size_t ws_size, hipStream_t stream) {
  const float* x       = (const float*)d_in[0];
  const int*   edge    = (const int*)d_in[1];
  const float* eps_z   = (const float*)d_in[3];
  const float* eps_gat = (const float*)d_in[4];
  const float* enc_W0  = (const float*)d_in[5];
  const float* enc_b0  = (const float*)d_in[6];
  const float* enc_W1  = (const float*)d_in[7];
  const float* enc_b1  = (const float*)d_in[8];
  const float* enc_Wm  = (const float*)d_in[9];
  const float* enc_bm  = (const float*)d_in[10];
  const float* enc_Wv  = (const float*)d_in[11];
  const float* enc_bv  = (const float*)d_in[12];
  const float* gm_Wl   = (const float*)d_in[13];
  const float* gm_bl   = (const float*)d_in[14];
  const float* gm_Wr   = (const float*)d_in[15];
  const float* gm_br   = (const float*)d_in[16];
  const float* gm_att  = (const float*)d_in[17];
  const float* gm_bias = (const float*)d_in[18];
  const float* gv_Wl   = (const float*)d_in[19];
  const float* gv_bl   = (const float*)d_in[20];
  const float* gv_Wr   = (const float*)d_in[21];
  const float* gv_br   = (const float*)d_in[22];
  const float* gv_att  = (const float*)d_in[23];
  const float* gv_bias = (const float*)d_in[24];
  const float* dec_W0  = (const float*)d_in[25];
  const float* dec_b0  = (const float*)d_in[26];
  const float* dec_Ws  = (const float*)d_in[27];
  const float* dec_bs  = (const float*)d_in[28];
  float* out = (float*)d_out;

  const int* e_src = edge;
  const int* e_dst = edge + EE;

  // Workspace carve-up (256B aligned).
  char* ws = (char*)d_ws;
  size_t off = 0;
  auto take = [&](size_t bytes) -> char* {
    size_t a = (off + 255) & ~(size_t)255;
    off = a + bytes;
    return ws + a;
  };
  _Float16* pkW0 = (_Float16*)take((size_t)63 * 8 * 512 * 2);    // 63 kchunks x 8 ntiles
  _Float16* pkW1 = (_Float16*)take((size_t)4 * 8 * 512 * 2);
  _Float16* pkWs = (_Float16*)take((size_t)4 * 125 * 512 * 2);
  float* h1      = (float*)take((size_t)NN * HH * 4);            // reused as hd later
  float* h2      = (float*)take((size_t)NN * HH * 4);
  float* libsum  = (float*)take((size_t)NN * 4);
  float* z       = (float*)take((size_t)NN * ZZ * 4);
  float* feat    = (float*)take((size_t)NN * SS * 4);
  float* xlm     = (float*)take((size_t)NN * SS * 4);
  float* xrm     = (float*)take((size_t)NN * SS * 4);
  float* xlv     = (float*)take((size_t)NN * SS * 4);
  float* xrv     = (float*)take((size_t)NN * SS * 4);
  float* e_m     = (float*)take((size_t)EE * 4);
  float* e_v     = (float*)take((size_t)EE * 4);
  float* m_m     = (float*)take((size_t)NN * 4);
  float* m_v     = (float*)take((size_t)NN * 4);
  float* s_m     = (float*)take((size_t)NN * 4);
  float* s_v     = (float*)take((size_t)NN * 4);
  float* acc_m   = (float*)take((size_t)NN * SS * 4);
  float* acc_v   = (float*)take((size_t)NN * SS * 4);
  float* hd      = h1;  // h1 is dead after enc1 -> reuse for decoder input
  (void)ws_size; (void)in_sizes; (void)n_in; (void)out_size;

  // 1) pack weights into WMMA B layout (f16)
  {
    int tot0 = 63 * 8 * 512;
    pack_b_kernel<<<(tot0 + 255) / 256, 256, 0, stream>>>(enc_W0, pkW0, GG, HH, 8, tot0);
    int tot1 = 4 * 8 * 512;
    pack_b_kernel<<<(tot1 + 255) / 256, 256, 0, stream>>>(enc_W1, pkW1, HH, HH, 8, tot1);
    int tot2 = 4 * 125 * 512;
    pack_b_kernel<<<(tot2 + 255) / 256, 256, 0, stream>>>(dec_Ws, pkWs, HH, GG, 125, tot2);
  }

  // 2) init GAT segment buffers (must happen every launch)
  gat_init_kernel<<<(NN * SS + 255) / 256, 256, 0, stream>>>(m_m, m_v, s_m, s_v,
                                                             acc_m, acc_v);

  // 3) encoder (WMMA) + library fusion
  enc0_kernel<<<(NTILES + 3) / 4, 128, 0, stream>>>(x, pkW0, enc_b0, h1, libsum);
  enc1_kernel<<<(NTILES + 3) / 4, 128, 0, stream>>>(h1, pkW1, enc_b1, h2);

  // 4) VAE head: q_m, q_v, z, feat
  vae_head_kernel<<<(NN + 3) / 4, 128, 0, stream>>>(h2, enc_Wm, enc_bm, enc_Wv,
                                                    enc_bv, eps_z, z, feat);

  // 5) GATv2 (mean & var branches), 3 edge passes with segment atomics
  gat_linear_kernel<<<(NN + 255) / 256, 256, 0, stream>>>(
      feat, gm_Wl, gm_bl, gm_Wr, gm_br, gv_Wl, gv_bl, gv_Wr, gv_br, xlm, xrm,
      xlv, xrv);
  edge_score_kernel<<<(EE + 255) / 256, 256, 0, stream>>>(
      e_src, e_dst, xlm, xrm, xlv, xrv, gm_att, gv_att, e_m, e_v, m_m, m_v);
  edge_exp_kernel<<<(EE + 255) / 256, 256, 0, stream>>>(e_dst, e_m, e_v, m_m,
                                                        m_v, s_m, s_v);
  edge_accum_kernel<<<(EE + 255) / 256, 256, 0, stream>>>(
      e_src, e_dst, e_m, e_v, s_m, s_v, xlm, xlv, acc_m, acc_v);

  // 6) node head: z_gat, z_all, dec layer0 + layernorm + relu -> hd
  node_head_kernel<<<(NN + 3) / 4, 128, 0, stream>>>(
      acc_m, acc_v, gm_bias, gv_bias, eps_gat, z, dec_W0, dec_b0, hd);

  // 7) decoder GEMM (WMMA) -> logits in d_out, then softmax * rowsum scale
  dec_gemm_kernel<<<dim3((NTILES + 3) / 4, 5), 128, 0, stream>>>(hd, pkWs,
                                                                 dec_bs, out);
  softmax_kernel<<<NN, 256, 0, stream>>>(out, libsum);
}
